// ToxicityGNN_2808908612250
// MI455X (gfx1250) — matrix operand order repaired and checked by reading, not compiled, not run on hardware
//
#include <hip/hip_runtime.h>
#include <hip/hip_bf16.h>

typedef __attribute__((ext_vector_type(16))) _Float16 v16h;
typedef __attribute__((ext_vector_type(8)))  _Float16 h8;
typedef __attribute__((ext_vector_type(8)))  float    v8f;

constexpr int N_NODES = 100000;
constexpr int N_EDGES = 400000;
constexpr int E_TOT   = N_EDGES + N_NODES;   // edges + self loops
constexpr int F_IN = 128;
constexpr int H    = 4;
constexpr int C1   = 64;
constexpr int C2   = 128;
constexpr int D1   = H * C1;   // 256
constexpr int D2   = H * C2;   // 512
constexpr int NG   = 2048;
constexpr int FPW  = 2048;
constexpr int HID  = 64;
constexpr int ZW   = D2 + HID; // 576
constexpr float NEG_SLOPE = 0.2f;

// ---------- monotonic float<->uint mapping for atomic float max ----------
__device__ __forceinline__ unsigned f2ord(float f) {
  unsigned u = __float_as_uint(f);
  return (u & 0x80000000u) ? ~u : (u | 0x80000000u);
}
__device__ __forceinline__ float ord2f(unsigned m) {
  return (m & 0x80000000u) ? __uint_as_float(m & 0x7fffffffu) : __uint_as_float(~m);
}

// ---------- small utility kernels ----------
__global__ void k_f32_to_f16(const float* __restrict__ in, _Float16* __restrict__ out, size_t n) {
  size_t i = blockIdx.x * (size_t)blockDim.x + threadIdx.x;
  if (i < n) out[i] = (_Float16)in[i];
}
// in[K,N] (f32, row-major)  ->  out[N,K] (f16)
__global__ void k_convert_T(const float* __restrict__ in, _Float16* __restrict__ out, int K, int N) {
  int i = blockIdx.x * blockDim.x + threadIdx.x;
  if (i >= K * N) return;
  int k = i / N, n = i % N;
  out[(size_t)n * K + k] = (_Float16)in[i];
}
__global__ void k_fill_f32(float* __restrict__ p, float v, size_t n) {
  size_t i = blockIdx.x * (size_t)blockDim.x + threadIdx.x;
  if (i < n) p[i] = v;
}
__global__ void k_fill_u32(unsigned* __restrict__ p, unsigned v, size_t n) {
  size_t i = blockIdx.x * (size_t)blockDim.x + threadIdx.x;
  if (i < n) p[i] = v;
}

// ---------- WMMA GEMM: C[M,N] = A[M,K] * Bt[N,K]^T, f16 in / f32 accumulate ----
// One wave computes a 16x64 strip (4 ntiles) so one A fragment feeds 4 WMMAs.
// A fragment: two contiguous 16B loads per k-step (lane half
// hs holds K = hs*16+{0..15}, matching B).  B fragment: 16 contiguous halves
// of a Bt row per lane half (K = hs*16+{0..15}) -> one 32B vector load.
template <typename OutT>
__global__ void k_wmma_gemm(const _Float16* __restrict__ A, const _Float16* __restrict__ Bt,
                            OutT* __restrict__ C, int M, int N, int K) {
  const int lane = threadIdx.x & 31;
  const int wave = threadIdx.x >> 5;
  const int m0   = blockIdx.x * 16;
  const int ng   = blockIdx.y * 8 + wave;     // group of 4 consecutive ntiles
  if (ng * 64 >= N) return;                   // wave-uniform exit
  const int r  = lane & 15;
  const int hs = lane >> 4;

  const _Float16* ap = A + (size_t)(m0 + r) * K + hs * 16;
  const _Float16* bp0 = Bt + (size_t)(ng * 64 +  0 + r) * K + hs * 16;
  const _Float16* bp1 = Bt + (size_t)(ng * 64 + 16 + r) * K + hs * 16;
  const _Float16* bp2 = Bt + (size_t)(ng * 64 + 32 + r) * K + hs * 16;
  const _Float16* bp3 = Bt + (size_t)(ng * 64 + 48 + r) * K + hs * 16;

  v8f acc[4] = {};
  for (int kk = 0; kk < K; kk += 32) {
    h8 alo = *(const h8*)(ap + kk);
    h8 ahi = *(const h8*)(ap + kk + 8);
    v16h a = __builtin_shufflevector(alo, ahi, 0, 1, 2, 3, 4, 5, 6, 7,
                                                8, 9, 10, 11, 12, 13, 14, 15);
    v16h b0 = *(const v16h*)(bp0 + kk);
    v16h b1 = *(const v16h*)(bp1 + kk);
    v16h b2 = *(const v16h*)(bp2 + kk);
    v16h b3 = *(const v16h*)(bp3 + kk);
    acc[0] = __builtin_amdgcn_wmma_f32_16x16x32_f16(false, a, false, b0, (short)0, acc[0], false, false);
    acc[1] = __builtin_amdgcn_wmma_f32_16x16x32_f16(false, a, false, b1, (short)0, acc[1], false, false);
    acc[2] = __builtin_amdgcn_wmma_f32_16x16x32_f16(false, a, false, b2, (short)0, acc[2], false, false);
    acc[3] = __builtin_amdgcn_wmma_f32_16x16x32_f16(false, a, false, b3, (short)0, acc[3], false, false);
  }
#pragma unroll
  for (int t = 0; t < 4; ++t) {
    OutT* cp = C + (size_t)(m0 + hs * 8) * N + ng * 64 + t * 16 + r;
#pragma unroll
    for (int v = 0; v < 8; ++v) cp[(size_t)v * N] = (OutT)acc[t][v];
  }
}

// ---------- per-node attention scores: e = sum_c h[n,hd,c]*a[hd,c] ----------
__global__ void k_attn_scores(const _Float16* __restrict__ h, const float* __restrict__ a_src,
                              const float* __restrict__ a_dst, float* __restrict__ es,
                              float* __restrict__ ed, int n, int ch) {
  int i = blockIdx.x * blockDim.x + threadIdx.x;
  if (i >= n * H) return;
  int node = i / H, hd = i % H;
  const _Float16* hp = h + (size_t)node * H * ch + hd * ch;
  const float* as = a_src + hd * ch;
  const float* ad = a_dst + hd * ch;
  float s0 = 0.f, s1 = 0.f;
  for (int c = 0; c < ch; c += 8) {
    h8 v = *(const h8*)(hp + c);
#pragma unroll
    for (int q = 0; q < 8; ++q) {
      float f = (float)v[q];
      s0 += f * as[c + q];
      s1 += f * ad[c + q];
    }
  }
  es[i] = s0;
  ed[i] = s1;
}

// ---------- edge pass 1: leaky-relu logits + per-dst atomic max ----------
__global__ void k_edge_logits_max(const int* __restrict__ ei, const float* __restrict__ es,
                                  const float* __restrict__ ed, float* __restrict__ logits,
                                  unsigned* __restrict__ maxv) {
  int e = blockIdx.x * blockDim.x + threadIdx.x;
  if (e >= E_TOT) return;
  int s = (e < N_EDGES) ? ei[e]           : (e - N_EDGES);
  int d = (e < N_EDGES) ? ei[N_EDGES + e] : (e - N_EDGES);
#pragma unroll
  for (int hd = 0; hd < H; ++hd) {
    float l = es[s * H + hd] + ed[d * H + hd];
    l = (l > 0.f) ? l : NEG_SLOPE * l;
    logits[(size_t)e * H + hd] = l;
    atomicMax(&maxv[d * H + hd], f2ord(l));
  }
}

// ---------- edge pass 2: p = exp(l - max), per-dst sum ----------
__global__ void k_edge_expsum(const int* __restrict__ ei, float* __restrict__ logits,
                              const unsigned* __restrict__ maxv, float* __restrict__ denom) {
  int e = blockIdx.x * blockDim.x + threadIdx.x;
  if (e >= E_TOT) return;
  int d = (e < N_EDGES) ? ei[N_EDGES + e] : (e - N_EDGES);
#pragma unroll
  for (int hd = 0; hd < H; ++hd) {
    float p = expf(logits[(size_t)e * H + hd] - ord2f(maxv[d * H + hd]));
    logits[(size_t)e * H + hd] = p;   // reuse buffer to hold p
    atomicAdd(&denom[d * H + hd], p);
  }
}

// ---------- edge pass 3: out[d] += alpha * h[s] (b128 gather, f32 scatter) ----
template <int CH>
__global__ void k_edge_aggregate(const int* __restrict__ ei, const float* __restrict__ p,
                                 const float* __restrict__ denom,
                                 const _Float16* __restrict__ h, float* __restrict__ out) {
  constexpr int C   = H * CH;
  constexpr int TPE = C / 8;        // threads per edge, each handles 8 channels
  constexpr int EPB = 256 / TPE;    // edges per block
  int e = blockIdx.x * EPB + (int)threadIdx.x / TPE;
  if (e >= E_TOT) return;
  int j  = (int)threadIdx.x % TPE;
  int s = (e < N_EDGES) ? ei[e]           : (e - N_EDGES);
  int d = (e < N_EDGES) ? ei[N_EDGES + e] : (e - N_EDGES);
  int c0 = j * 8;
  int hd = c0 / CH;                 // CH >= 8, so one head per 8-chunk
  float alpha = p[(size_t)e * H + hd] / denom[d * H + hd];
  h8 m = *(const h8*)(h + (size_t)s * C + c0);
  float* op = out + (size_t)d * C + c0;
#pragma unroll
  for (int q = 0; q < 8; ++q) atomicAdd(op + q, alpha * (float)m[q]);
}

// ---------- bias + elu, write fp32 and/or f16 ----------
__global__ void k_bias_elu(const float* __restrict__ in, const float* __restrict__ bias,
                           float* __restrict__ out32, _Float16* __restrict__ out16,
                           size_t n, int C) {
  size_t i = blockIdx.x * (size_t)blockDim.x + threadIdx.x;
  if (i >= n) return;
  float v = in[i] + bias[(int)(i % C)];
  v = (v > 0.f) ? v : (expf(v) - 1.f);
  if (out32) out32[i] = v;
  if (out16) out16[i] = (_Float16)v;
}

// ---------- global mean pool (accumulate, 4-wide) ----------
__global__ void k_pool(const float* __restrict__ act, const int* __restrict__ batch,
                       float* __restrict__ sums, float* __restrict__ counts, int n, int C) {
  size_t i = blockIdx.x * (size_t)blockDim.x + threadIdx.x;
  size_t total = (size_t)n * (C / 4);
  if (i >= total) return;
  int node = (int)(i / (C / 4));
  int c4   = (int)(i % (C / 4)) * 4;
  int g = batch[node];
  const float4 v = *(const float4*)(act + (size_t)node * C + c4);
  float* sp = sums + (size_t)g * C + c4;
  atomicAdd(sp + 0, v.x);
  atomicAdd(sp + 1, v.y);
  atomicAdd(sp + 2, v.z);
  atomicAdd(sp + 3, v.w);
  if (c4 == 0) atomicAdd(&counts[g], 1.0f);
}

// ---------- z = concat(pooled, fp@Wfp + bfp) as f16 ----------
__global__ void k_build_z(const float* __restrict__ sums, const float* __restrict__ counts,
                          const float* __restrict__ fph, const float* __restrict__ bfp,
                          _Float16* __restrict__ z) {
  int i = blockIdx.x * blockDim.x + threadIdx.x;
  if (i >= NG * ZW) return;
  int g = i / ZW, c = i % ZW;
  float v;
  if (c < D2) v = sums[(size_t)g * D2 + c] / fmaxf(counts[g], 1.0f);
  else        v = fph[(size_t)g * HID + (c - D2)] + bfp[c - D2];
  z[i] = (_Float16)v;
}

// ---------- fc1 relu+bias fused with fc2 dot ----------
__global__ void k_fc2(const float* __restrict__ z2raw, const float* __restrict__ bfc1,
                      const float* __restrict__ Wfc2, const float* __restrict__ bfc2,
                      float* __restrict__ out) {
  int g = blockIdx.x * blockDim.x + threadIdx.x;
  if (g >= NG) return;
  float acc = bfc2[0];
  for (int c = 0; c < HID; ++c) {
    float v = z2raw[(size_t)g * HID + c] + bfc1[c];
    v = (v > 0.f) ? v : 0.f;
    acc += v * Wfc2[c];
  }
  out[g] = acc;
}

static inline unsigned nblk(size_t n, int bs) { return (unsigned)((n + bs - 1) / bs); }

extern "C" void kernel_launch(void* const* d_in, const int* in_sizes, int n_in,
                              void* d_out, int out_size, void* d_ws, size_t ws_size,
                              hipStream_t stream) {
  const float* x      = (const float*)d_in[0];
  const int*   ei     = (const int*)  d_in[1];
  const int*   batch  = (const int*)  d_in[2];
  const float* fp     = (const float*)d_in[3];
  const float* W1     = (const float*)d_in[4];
  const float* a_src1 = (const float*)d_in[5];
  const float* a_dst1 = (const float*)d_in[6];
  const float* b1     = (const float*)d_in[7];
  const float* W2     = (const float*)d_in[8];
  const float* a_src2 = (const float*)d_in[9];
  const float* a_dst2 = (const float*)d_in[10];
  const float* b2     = (const float*)d_in[11];
  const float* Wfp    = (const float*)d_in[12];
  const float* bfp    = (const float*)d_in[13];
  const float* Wfc1   = (const float*)d_in[14];
  const float* bfc1   = (const float*)d_in[15];
  const float* Wfc2   = (const float*)d_in[16];
  const float* bfc2   = (const float*)d_in[17];
  float* out = (float*)d_out;

  // bump allocator over workspace
  char* w = (char*)d_ws;
  size_t off = 0;
  auto alloc = [&](size_t bytes) -> void* {
    void* p = w + off;
    off += (bytes + 255) & ~(size_t)255;
    return p;
  };

  _Float16* x16    = (_Float16*)alloc((size_t)N_NODES * F_IN * 2);
  _Float16* W1t    = (_Float16*)alloc((size_t)D1 * F_IN * 2);     // [D1, F_IN]
  _Float16* W2t    = (_Float16*)alloc((size_t)D2 * D1 * 2);       // [D2, D1]
  _Float16* fph16  = (_Float16*)alloc((size_t)NG * FPW * 2);
  _Float16* Wfpt   = (_Float16*)alloc((size_t)HID * FPW * 2);     // [HID, FPW]
  _Float16* Wfc1t  = (_Float16*)alloc((size_t)HID * ZW * 2);      // [HID, ZW]
  _Float16* hbuf   = (_Float16*)alloc((size_t)N_NODES * D2 * 2);  // h1 then h2 (f16)
  _Float16* act16  = (_Float16*)alloc((size_t)N_NODES * D1 * 2);  // elu(out1) f16
  float*    outbuf = (float*)   alloc((size_t)N_NODES * D2 * 4);  // out1 then out2
  float*    es     = (float*)   alloc((size_t)N_NODES * H * 4);
  float*    ed     = (float*)   alloc((size_t)N_NODES * H * 4);
  unsigned* maxv   = (unsigned*)alloc((size_t)N_NODES * H * 4);
  float*    denom  = (float*)   alloc((size_t)N_NODES * H * 4);
  float*    logits = (float*)   alloc((size_t)E_TOT * H * 4);
  float*    sums   = (float*)   alloc((size_t)NG * D2 * 4);
  float*    counts = (float*)   alloc((size_t)NG * 4);
  float*    fphr   = (float*)   alloc((size_t)NG * HID * 4);
  _Float16* z16    = (_Float16*)alloc((size_t)NG * ZW * 2);
  float*    z2raw  = (float*)   alloc((size_t)NG * HID * 4);
  (void)ws_size; (void)in_sizes; (void)n_in; (void)out_size;

  const int BS = 256;
  const unsigned ORD_NEG_INF = 0x007FFFFFu;  // f2ord(-inf)

  // ---- f16 conversions / weight transposes ----
  k_f32_to_f16<<<nblk((size_t)N_NODES * F_IN, BS), BS, 0, stream>>>(x, x16, (size_t)N_NODES * F_IN);
  k_f32_to_f16<<<nblk((size_t)NG * FPW, BS), BS, 0, stream>>>(fp, fph16, (size_t)NG * FPW);
  k_convert_T<<<nblk((size_t)F_IN * D1, BS), BS, 0, stream>>>(W1, W1t, F_IN, D1);
  k_convert_T<<<nblk((size_t)D1 * D2, BS), BS, 0, stream>>>(W2, W2t, D1, D2);
  k_convert_T<<<nblk((size_t)FPW * HID, BS), BS, 0, stream>>>(Wfp, Wfpt, FPW, HID);
  k_convert_T<<<nblk((size_t)ZW * HID, BS), BS, 0, stream>>>(Wfc1, Wfc1t, ZW, HID);

  // ================= GAT layer 1 =================
  {
    dim3 grid(N_NODES / 16, (unsigned)((D1 + 511) / 512));
    k_wmma_gemm<_Float16><<<grid, 256, 0, stream>>>(x16, W1t, hbuf, N_NODES, D1, F_IN);
  }
  k_attn_scores<<<nblk((size_t)N_NODES * H, BS), BS, 0, stream>>>(hbuf, a_src1, a_dst1, es, ed, N_NODES, C1);
  k_fill_u32<<<nblk((size_t)N_NODES * H, BS), BS, 0, stream>>>(maxv, ORD_NEG_INF, (size_t)N_NODES * H);
  k_fill_f32<<<nblk((size_t)N_NODES * H, BS), BS, 0, stream>>>(denom, 0.f, (size_t)N_NODES * H);
  k_fill_f32<<<nblk((size_t)N_NODES * D1, BS), BS, 0, stream>>>(outbuf, 0.f, (size_t)N_NODES * D1);
  k_edge_logits_max<<<nblk(E_TOT, BS), BS, 0, stream>>>(ei, es, ed, logits, maxv);
  k_edge_expsum<<<nblk(E_TOT, BS), BS, 0, stream>>>(ei, logits, maxv, denom);
  k_edge_aggregate<C1><<<nblk(E_TOT, 256 / (H * C1 / 8)), 256, 0, stream>>>(ei, logits, denom, hbuf, outbuf);
  k_bias_elu<<<nblk((size_t)N_NODES * D1, BS), BS, 0, stream>>>(outbuf, b1, (float*)nullptr, act16,
                                                               (size_t)N_NODES * D1, D1);

  // ================= GAT layer 2 =================
  {
    dim3 grid(N_NODES / 16, (unsigned)((D2 + 511) / 512));
    k_wmma_gemm<_Float16><<<grid, 256, 0, stream>>>(act16, W2t, hbuf, N_NODES, D2, D1);
  }
  k_attn_scores<<<nblk((size_t)N_NODES * H, BS), BS, 0, stream>>>(hbuf, a_src2, a_dst2, es, ed, N_NODES, C2);
  k_fill_u32<<<nblk((size_t)N_NODES * H, BS), BS, 0, stream>>>(maxv, ORD_NEG_INF, (size_t)N_NODES * H);
  k_fill_f32<<<nblk((size_t)N_NODES * H, BS), BS, 0, stream>>>(denom, 0.f, (size_t)N_NODES * H);
  k_fill_f32<<<nblk((size_t)N_NODES * D2, BS), BS, 0, stream>>>(outbuf, 0.f, (size_t)N_NODES * D2);
  k_edge_logits_max<<<nblk(E_TOT, BS), BS, 0, stream>>>(ei, es, ed, logits, maxv);
  k_edge_expsum<<<nblk(E_TOT, BS), BS, 0, stream>>>(ei, logits, maxv, denom);
  k_edge_aggregate<C2><<<nblk(E_TOT, 256 / (H * C2 / 8)), 256, 0, stream>>>(ei, logits, denom, hbuf, outbuf);
  k_bias_elu<<<nblk((size_t)N_NODES * D2, BS), BS, 0, stream>>>(outbuf, b2, outbuf, (_Float16*)nullptr,
                                                               (size_t)N_NODES * D2, D2);

  // ================= pooling + fingerprint + head =================
  k_fill_f32<<<nblk((size_t)NG * D2, BS), BS, 0, stream>>>(sums, 0.f, (size_t)NG * D2);
  k_fill_f32<<<nblk((size_t)NG, BS), BS, 0, stream>>>(counts, 0.f, (size_t)NG);
  k_pool<<<nblk((size_t)N_NODES * (D2 / 4), BS), BS, 0, stream>>>(outbuf, batch, sums, counts, N_NODES, D2);

  {
    dim3 grid(NG / 16, 1);   // HID=64 -> one 4-ntile group, wave 0 only (guarded)
    k_wmma_gemm<float><<<grid, 256, 0, stream>>>(fph16, Wfpt, fphr, NG, HID, FPW);
  }
  k_build_z<<<nblk((size_t)NG * ZW, BS), BS, 0, stream>>>(sums, counts, fphr, bfp, z16);

  {
    dim3 grid(NG / 16, 1);
    k_wmma_gemm<float><<<grid, 256, 0, stream>>>(z16, Wfc1t, z2raw, NG, HID, ZW);
  }
  k_fc2<<<nblk((size_t)NG, BS), BS, 0, stream>>>(z2raw, bfc1, Wfc2, bfc2, out);
}
